// DynSMHALayer_16853451670032
// MI455X (gfx1250) — compile-verified, matrix operand order checked by reading, not run on hardware
//
#include <hip/hip_runtime.h>
#include <hip/hip_bf16.h>

typedef __bf16 bf16_t;
typedef __attribute__((ext_vector_type(16))) __bf16 v16bf;
typedef __attribute__((ext_vector_type(8)))  __bf16 v8bf;
typedef __attribute__((ext_vector_type(8)))  float  v8f;
typedef __attribute__((ext_vector_type(4)))  int    v4i;

#define BB 4
#define TT 4096
#define CC 1024
#define WW 64
#define EE 16
#define NN 64            // TT / WW
#define DD 64
#define KCOMP (WW * CC)  // 65536
#define NEGV -1000000000.0f

// ---------- gfx1250 async global->LDS copy (ASYNCcnt path), guarded ----------
#if defined(__has_builtin)
#  if __has_builtin(__builtin_amdgcn_global_load_async_to_lds_b128)
#    define HAVE_ASYNC_B128 1
#  endif
#endif
#ifndef HAVE_ASYNC_B128
#  define HAVE_ASYNC_B128 0
#endif

typedef __attribute__((address_space(1))) v4i as1_v4i;  // global (printed __device__)
typedef __attribute__((address_space(3))) v4i as3_v4i;  // LDS

__device__ __forceinline__ void copy16_async_or_direct(const void* g, void* l) {
#if HAVE_ASYNC_B128
  __builtin_amdgcn_global_load_async_to_lds_b128(
      (as1_v4i*)(unsigned long long)(size_t)g,
      (as3_v4i*)(unsigned)(size_t)l, 0, 0);
#else
  *(uint4*)l = *(const uint4*)g;
#endif
}
__device__ __forceinline__ void async_wait0() {
#if HAVE_ASYNC_B128
#  if __has_builtin(__builtin_amdgcn_s_wait_asynccnt)
  __builtin_amdgcn_s_wait_asynccnt(0);
#  else
  asm volatile("s_wait_asynccnt 0" ::: "memory");
#  endif
#endif
}

// ---------- bf16 helper (bit-exact RNE) -------------------------------------
__device__ __forceinline__ bf16_t f2bf(float f) {
  union { float f; unsigned u; } x; x.f = f;
  unsigned r = (x.u + 0x7FFFu + ((x.u >> 16) & 1u)) >> 16;
  unsigned short h = (unsigned short)r;
  bf16_t o; __builtin_memcpy(&o, &h, 2); return o;
}
__device__ __forceinline__ v8bf cvt8(float4 a, float4 b) {
  v8bf v;
  v[0] = f2bf(a.x); v[1] = f2bf(a.y); v[2] = f2bf(a.z); v[3] = f2bf(a.w);
  v[4] = f2bf(b.x); v[5] = f2bf(b.y); v[6] = f2bf(b.z); v[7] = f2bf(b.w);
  return v;
}

// ---------- packed WMMA fragment load: 2x ds_load_b128 + shuffle -------------
// All LDS tiles are stored K-major packed: every fragment (A, B, K^T, P, V^T)
// is two contiguous 16B reads per lane.
// f[0..7] = K kh..kh+7, f[8..15] = K 16+kh..16+kh+7, kh=(lane>>4)*8, r=lane&15.
__device__ __forceinline__ v16bf frag_packed(const bf16_t* base, int ld, int row0,
                                             int k0, int lane) {
  int r  = row0 + (lane & 15);
  int kh = (lane >> 4) * 8;
  const bf16_t* p = base + (size_t)r * ld + k0 + kh;
  v8bf lo = *(const v8bf*)(p);
  v8bf hi = *(const v8bf*)(p + 16);
  return __builtin_shufflevector(lo, hi, 0, 1, 2, 3, 4, 5, 6, 7,
                                 8, 9, 10, 11, 12, 13, 14, 15);
}
__device__ __forceinline__ v8f wmma_bf16(v16bf a, v16bf b, v8f c) {
  return __builtin_amdgcn_wmma_f32_16x16x32_bf16(false, a, false, b, (short)0, c,
                                                 false, false);
}

// ============================================================================
// 1) compress GEMM: M=256, K=65536, N=1024. 64x64 tile, double-buffered LDS.
// ============================================================================
__global__ void __launch_bounds__(256) compress_kernel(
    const float* __restrict__ hs, const float* __restrict__ wc,
    const float* __restrict__ bias, float* __restrict__ comp) {
  __shared__ __align__(16) bf16_t As[2][64 * 32];
  __shared__ __align__(16) bf16_t Bs[2][64 * 32];  // packed [n][k]
  const int n0 = blockIdx.x * 64;
  const int m0 = blockIdx.y * 64;
  const int t  = threadIdx.x;
  const int wv = t >> 5, lane = t & 31;
  const int wm = wv >> 1, wn = wv & 1;
  const int r = t >> 2, seg = (t & 3) * 8;  // 64 rows x 4 segs of 8
  const float* aRow = hs + (size_t)(m0 + r) * KCOMP + seg;
  const float* bRow = wc + (size_t)(n0 + r) * KCOMP + seg;

  auto stage = [&](int buf, int k0) {
    const float4* a4 = (const float4*)(aRow + k0);
    const float4* b4 = (const float4*)(bRow + k0);
    float4 a0 = a4[0], a1 = a4[1];
    float4 b0 = b4[0], b1 = b4[1];
    *(v8bf*)&As[buf][r * 32 + seg] = cvt8(a0, a1);
    *(v8bf*)&Bs[buf][r * 32 + seg] = cvt8(b0, b1);
  };

  stage(0, 0);
  v8f acc0 = {}, acc1 = {};
  int buf = 0;
  for (int k0 = 0; k0 < KCOMP; k0 += 32) {
    __syncthreads();
    if (k0 + 32 < KCOMP) {
      stage(buf ^ 1, k0 + 32);
      __builtin_prefetch(bRow + k0 + 64, 0, 1);  // global_prefetch into L2
    }
    v16bf a  = frag_packed(As[buf], 32, wm * 16, 0, lane);
    v16bf b0 = frag_packed(Bs[buf], 32, wn * 32 + 0, 0, lane);
    v16bf b1 = frag_packed(Bs[buf], 32, wn * 32 + 16, 0, lane);
    acc0 = wmma_bf16(a, b0, acc0);
    acc1 = wmma_bf16(a, b1, acc1);
    buf ^= 1;
  }
  const int hi = lane >> 4, cl = lane & 15;
#pragma unroll
  for (int rr = 0; rr < 8; ++rr) {
    int rowg = m0 + wm * 16 + rr + hi * 8;
    int c0   = n0 + wn * 32 + cl;
    comp[(size_t)rowg * CC + c0]      = fmaxf(acc0[rr] + bias[c0], 0.f);
    comp[(size_t)rowg * CC + c0 + 16] = fmaxf(acc1[rr] + bias[c0 + 16], 0.f);
  }
}

// ============================================================================
// 2) column rsqrt-norms of g_sim (t<16) and f_sim (t>=16): rsn[32]
// ============================================================================
__global__ void colnorm_kernel(const float* __restrict__ g_sim,
                               const float* __restrict__ f_sim,
                               float* __restrict__ rsn) {
  int t = threadIdx.x;
  const float* sim = (t < 16) ? g_sim : f_sim;
  int e = t & 15;
  float s = 0.f;
  for (int c = 0; c < CC; ++c) { float v = sim[c * EE + e]; s += v * v; }
  rsn[t] = rsqrtf(s + 1e-12f);
}

// ============================================================================
// 3) gating (block-level mode 0 / token-level mode 1)
// ============================================================================
__global__ void __launch_bounds__(256) gate_kernel(
    const float* __restrict__ x, const float* __restrict__ sim,
    const float* __restrict__ rsn, const float* __restrict__ gates,
    const float* __restrict__ thr, const float* __restrict__ bmask_in,
    float* __restrict__ bmask_out, float* __restrict__ rw_out, int mode) {
  __shared__ float red[256];
  __shared__ float slog[EE];
  __shared__ float sprob[EE];
  const int row = blockIdx.x;
  const int t   = threadIdx.x;
  const float* xr = x + (size_t)row * CC;
  float x4[4];
#pragma unroll
  for (int i = 0; i < 4; ++i) x4[i] = xr[t + i * 256];
  float ss = 0.f;
#pragma unroll
  for (int i = 0; i < 4; ++i) ss += x4[i] * x4[i];
  red[t] = ss; __syncthreads();
  for (int s = 128; s > 0; s >>= 1) { if (t < s) red[t] += red[t + s]; __syncthreads(); }
  float rsx = rsqrtf(red[0] + 1e-12f); __syncthreads();
  for (int e = 0; e < EE; ++e) {
    float d = 0.f;
#pragma unroll
    for (int i = 0; i < 4; ++i) d += x4[i] * sim[(size_t)(t + i * 256) * EE + e];
    red[t] = d; __syncthreads();
    for (int s = 128; s > 0; s >>= 1) { if (t < s) red[t] += red[t + s]; __syncthreads(); }
    if (t == 0)
      slog[e] = red[0] * rsx * rsn[e] - 1.f / (1.f + __expf(-gates[e]));
    __syncthreads();
  }
  if (t == 0) {
    float gated[EE]; int act[EE]; int nact = 0;
    for (int e = 0; e < EE; ++e) {
      gated[e] = fmaxf(slog[e], 0.f);
      act[e]   = slog[e] > 0.f ? 1 : 0;
      nact    += act[e];
    }
    if (nact == 0) {  // top-2 fallback on raw logits
      int i1 = 0; float v1 = slog[0];
      for (int e = 1; e < EE; ++e) if (slog[e] > v1) { v1 = slog[e]; i1 = e; }
      int i2 = (i1 == 0) ? 1 : 0; float v2 = slog[i2];
      for (int e = 0; e < EE; ++e) if (e != i1 && slog[e] > v2) { v2 = slog[e]; i2 = e; }
      act[i1] = 1; act[i2] = 1;
    }
    float mx = NEGV; float msk[EE];
    for (int e = 0; e < EE; ++e) { msk[e] = act[e] ? gated[e] : NEGV; mx = fmaxf(mx, msk[e]); }
    float sum = 0.f;
    for (int e = 0; e < EE; ++e) { float ev = __expf(msk[e] - mx); sprob[e] = ev; sum += ev; }
    float inv = 1.f / sum;
    if (mode == 0) {
      float imp = 0.f;
      for (int e = 0; e < EE; ++e) imp = fmaxf(imp, sprob[e] * inv);
      bmask_out[row] = (imp - thr[0] > 0.f) ? 1.f : 0.f;
    } else {
      float m = bmask_in[row / WW];
      for (int e = 0; e < EE; ++e) rw_out[(size_t)row * EE + e] = sprob[e] * inv * m;
    }
  }
}

// ============================================================================
// 4) q/k/v projection GEMM (M=16384, K=1024, N tile = expert D=64) + RoPE.
// ============================================================================
__global__ void __launch_bounds__(256) proj_kernel(
    const float* __restrict__ hs, const float* __restrict__ wp,
    const int* __restrict__ pos_ids, bf16_t* __restrict__ out, int do_rope) {
  __shared__ __align__(16) bf16_t As[2][64 * 32];
  __shared__ __align__(16) bf16_t Bp[2][64 * 32];  // packed [d][k]
  __shared__ float Cs[64 * 64];
  const int ex = blockIdx.x;
  const int m0 = blockIdx.y * 64;
  const int t  = threadIdx.x;
  const int wv = t >> 5, lane = t & 31;
  const int wm = wv >> 1, wn = wv & 1;
  const int r = t >> 2, seg = (t & 3) * 8;
  const float* aRow = hs + (size_t)(m0 + r) * CC + seg;
  const float* wpe  = wp + (size_t)ex * CC * DD;

  auto stage = [&](int buf, int k0) {
    const float4* a4 = (const float4*)(aRow + k0);
    *(v8bf*)&As[buf][r * 32 + seg] = cvt8(a4[0], a4[1]);
#pragma unroll
    for (int i = 0; i < 2; ++i) {
      int g    = t + i * 256;          // 512 float4 groups
      int flat = g * 4;
      int k    = flat >> 6;            // 0..31
      int d0   = flat & 63;
      float4 w4 = *(const float4*)(wpe + (size_t)(k0 + k) * DD + d0);
      Bp[buf][(d0 + 0) * 32 + k] = f2bf(w4.x);
      Bp[buf][(d0 + 1) * 32 + k] = f2bf(w4.y);
      Bp[buf][(d0 + 2) * 32 + k] = f2bf(w4.z);
      Bp[buf][(d0 + 3) * 32 + k] = f2bf(w4.w);
    }
  };

  stage(0, 0);
  v8f acc0 = {}, acc1 = {};
  int buf = 0;
  for (int k0 = 0; k0 < CC; k0 += 32) {
    __syncthreads();
    if (k0 + 32 < CC) stage(buf ^ 1, k0 + 32);
    v16bf a  = frag_packed(As[buf], 32, wm * 16, 0, lane);
    v16bf b0 = frag_packed(Bp[buf], 32, wn * 32 + 0, 0, lane);
    v16bf b1 = frag_packed(Bp[buf], 32, wn * 32 + 16, 0, lane);
    acc0 = wmma_bf16(a, b0, acc0);
    acc1 = wmma_bf16(a, b1, acc1);
    buf ^= 1;
  }
  const int hi = lane >> 4, cl = lane & 15;
#pragma unroll
  for (int rr = 0; rr < 8; ++rr) {
    Cs[(wm * 16 + rr + hi * 8) * 64 + wn * 32 + cl]      = acc0[rr];
    Cs[(wm * 16 + rr + hi * 8) * 64 + wn * 32 + cl + 16] = acc1[rr];
  }
  __syncthreads();
#pragma unroll
  for (int i = 0; i < 16; ++i) {
    int e = t + i * 256;
    int rr = e >> 6, d = e & 63;
    int m = m0 + rr;
    float v = Cs[rr * 64 + d];
    if (do_rope) {
      int pos = pos_ids[m];
      int j   = d & 31;
      float invf = __expf(-(float)(2 * j) * (9.210340371976184f / 64.0f));
      float ang  = (float)pos * invf;
      float cv = __cosf(ang), sv = __sinf(ang);
      float rot = (d < 32) ? -Cs[rr * 64 + d + 32] : Cs[rr * 64 + d - 32];
      v = v * cv + rot * sv;
    }
    int b  = m >> 12;
    int tt = m & 4095;
    out[(((size_t)b * EE + ex) * TT + tt) * DD + d] = f2bf(v);
  }
}

// ============================================================================
// 5) per-(b,e,n) 64x64 causal attention. q/k tiles staged via async
//    global->LDS (ASYNCcnt); V staged transposed manually (needs transpose).
// ============================================================================
__global__ void __launch_bounds__(256) attn_kernel(
    const bf16_t* __restrict__ qb, const bf16_t* __restrict__ kb,
    const bf16_t* __restrict__ vb, const float* __restrict__ rw,
    bf16_t* __restrict__ ctxw) {
  __shared__ __align__(16) bf16_t qs[64 * 64];
  __shared__ __align__(16) bf16_t ks[64 * 64];
  __shared__ __align__(16) bf16_t vt[64 * 64];  // transposed [d][w']
  __shared__ __align__(16) bf16_t ps[64 * 64];
  __shared__ float ss[64 * 64];
  const int blk = blockIdx.x;
  const int n  = blk & 63;
  const int ex = (blk >> 6) & 15;
  const int b  = blk >> 10;
  const int t  = threadIdx.x;
  const size_t base = (((size_t)b * EE + ex) * TT + (size_t)n * WW) * DD;
  const uint4* gq = (const uint4*)(qb + base);
  const uint4* gk = (const uint4*)(kb + base);
  const uint4* gv = (const uint4*)(vb + base);
  uint4* lq = (uint4*)qs; uint4* lk = (uint4*)ks;
  // issue async q/k tile copies first ...
#pragma unroll
  for (int i = 0; i < 2; ++i) {
    int g = t + i * 256;
    copy16_async_or_direct(gq + g, lq + g);
    copy16_async_or_direct(gk + g, lk + g);
  }
  // ... overlap with the V transpose staging ...
#pragma unroll
  for (int i = 0; i < 2; ++i) {
    int g = t + i * 256;
    uint4 u = gv[g];                     // 8 consecutive d for one w'
    const bf16_t* pe = (const bf16_t*)&u;
    int flat = g * 8;
    int wp2  = flat >> 6;                // w' row
    int d0   = flat & 63;
#pragma unroll
    for (int j = 0; j < 8; ++j) vt[(d0 + j) * 64 + wp2] = pe[j];
  }
  async_wait0();       // ASYNCcnt -> 0 before making LDS visible
  __syncthreads();
  const int wv = t >> 5, lane = t & 31;
  const int wm = wv >> 1, wn = wv & 1;
  v8f s0 = {}, s1 = {};
#pragma unroll
  for (int kk = 0; kk < 64; kk += 32) {
    v16bf a  = frag_packed(qs, 64, wm * 16, kk, lane);
    v16bf b0 = frag_packed(ks, 64, wn * 32 + 0, kk, lane);   // K^T
    v16bf b1 = frag_packed(ks, 64, wn * 32 + 16, kk, lane);
    s0 = wmma_bf16(a, b0, s0);
    s1 = wmma_bf16(a, b1, s1);
  }
  const int hi = lane >> 4, cl = lane & 15;
  const float scale = 0.125f;  // 1/sqrt(D)
#pragma unroll
  for (int rr = 0; rr < 8; ++rr) {
    int row = wm * 16 + rr + hi * 8;
    ss[row * 64 + wn * 32 + cl]      = s0[rr] * scale;
    ss[row * 64 + wn * 32 + cl + 16] = s1[rr] * scale;
  }
  __syncthreads();
  if (t < 64) {  // causal softmax, one row per thread
    int rowi = t;
    float mx = -3.0e38f;
    for (int j = 0; j <= rowi; ++j) mx = fmaxf(mx, ss[rowi * 64 + j]);
    float sum = 0.f;
    for (int j = 0; j <= rowi; ++j) {
      float ev = __expf(ss[rowi * 64 + j] - mx);
      ss[rowi * 64 + j] = ev; sum += ev;
    }
    float inv = 1.f / sum;
    for (int j = 0; j < 64; ++j)
      ps[rowi * 64 + j] = f2bf(j <= rowi ? ss[rowi * 64 + j] * inv : 0.f);
  }
  __syncthreads();
  v8f c0 = {}, c1 = {};
#pragma unroll
  for (int kk = 0; kk < 64; kk += 32) {
    v16bf a  = frag_packed(ps, 64, wm * 16, kk, lane);
    v16bf b0 = frag_packed(vt, 64, wn * 32 + 0, kk, lane);   // V via transpose
    v16bf b1 = frag_packed(vt, 64, wn * 32 + 16, kk, lane);
    c0 = wmma_bf16(a, b0, c0);
    c1 = wmma_bf16(a, b1, c1);
  }
#pragma unroll
  for (int rr = 0; rr < 8; ++rr) {
    int wloc = wm * 16 + rr + hi * 8;
    int tt   = n * WW + wloc;
    float f  = rw[((size_t)b * TT + tt) * EE + ex];
    int d    = wn * 32 + cl;
    size_t o = ((size_t)b * TT + tt) * (EE * DD) + (size_t)ex * DD;
    ctxw[o + d]      = f2bf(c0[rr] * f);
    ctxw[o + d + 16] = f2bf(c1[rr] * f);
  }
}

// ============================================================================
// 6) output GEMM: out = ctxw(bf16) * o_proj. A tile staged via async
//    global->LDS (already bf16, contiguous); B staged transposed.
// ============================================================================
__global__ void __launch_bounds__(256) out_kernel(
    const bf16_t* __restrict__ aw, const float* __restrict__ wo,
    float* __restrict__ out) {
  __shared__ __align__(16) bf16_t As[2][64 * 32];
  __shared__ __align__(16) bf16_t Bp[2][64 * 32];  // packed [n][k]
  const int n0 = blockIdx.x * 64;
  const int m0 = blockIdx.y * 64;
  const int t  = threadIdx.x;
  const int wv = t >> 5, lane = t & 31;
  const int wm = wv >> 1, wn = wv & 1;
  const int r = t >> 2, seg = (t & 3) * 8;
  const bf16_t* aRow = aw + (size_t)(m0 + r) * (EE * DD) + seg;

  auto stage = [&](int buf, int k0) {
    // A already bf16: async 16B copy straight into LDS
    copy16_async_or_direct(aRow + k0, &As[buf][r * 32 + seg]);
    // B: read float4 along n (coalesced), scatter transposed into [n][k]
#pragma unroll
    for (int i = 0; i < 2; ++i) {
      int g    = t + i * 256;
      int flat = g * 4;
      int k    = flat >> 6;
      int nn   = flat & 63;
      float4 w4 = *(const float4*)(wo + (size_t)(k0 + k) * CC + n0 + nn);
      Bp[buf][(nn + 0) * 32 + k] = f2bf(w4.x);
      Bp[buf][(nn + 1) * 32 + k] = f2bf(w4.y);
      Bp[buf][(nn + 2) * 32 + k] = f2bf(w4.z);
      Bp[buf][(nn + 3) * 32 + k] = f2bf(w4.w);
    }
  };

  stage(0, 0);
  v8f acc0 = {}, acc1 = {};
  int buf = 0;
  for (int k0 = 0; k0 < EE * DD; k0 += 32) {
    async_wait0();       // my async A-tile writes landed
    __syncthreads();     // everyone's staging visible
    if (k0 + 32 < EE * DD) stage(buf ^ 1, k0 + 32);
    v16bf a  = frag_packed(As[buf], 32, wm * 16, 0, lane);
    v16bf b0 = frag_packed(Bp[buf], 32, wn * 32 + 0, 0, lane);
    v16bf b1 = frag_packed(Bp[buf], 32, wn * 32 + 16, 0, lane);
    acc0 = wmma_bf16(a, b0, acc0);
    acc1 = wmma_bf16(a, b1, acc1);
    buf ^= 1;
  }
  const int hi = lane >> 4, cl = lane & 15;
#pragma unroll
  for (int rr = 0; rr < 8; ++rr) {
    int rowg = m0 + wm * 16 + rr + hi * 8;
    int c0   = n0 + wn * 32 + cl;
    out[(size_t)rowg * CC + c0]      = acc0[rr];
    out[(size_t)rowg * CC + c0 + 16] = acc1[rr];
  }
}

// ============================================================================
extern "C" void kernel_launch(void* const* d_in, const int* in_sizes, int n_in,
                              void* d_out, int out_size, void* d_ws, size_t ws_size,
                              hipStream_t stream) {
  (void)in_sizes; (void)n_in; (void)out_size; (void)ws_size;
  const float* hs   = (const float*)d_in[0];
  const int*   pos  = (const int*)d_in[1];
  const float* wc   = (const float*)d_in[2];
  const float* bc   = (const float*)d_in[3];
  const float* gsim = (const float*)d_in[4];
  const float* ggat = (const float*)d_in[5];
  const float* thr  = (const float*)d_in[6];
  const float* fsim = (const float*)d_in[7];
  const float* fgat = (const float*)d_in[8];
  const float* qp   = (const float*)d_in[9];
  const float* kp   = (const float*)d_in[10];
  const float* vp   = (const float*)d_in[11];
  const float* op   = (const float*)d_in[12];

  char* ws = (char*)d_ws;
  const size_t szQKV = (size_t)BB * EE * TT * DD * sizeof(bf16_t);  // 32 MiB
  bf16_t* qb = (bf16_t*)(ws);
  bf16_t* kb = (bf16_t*)(ws + szQKV);
  bf16_t* vb = (bf16_t*)(ws + 2 * szQKV);
  bf16_t* cw = (bf16_t*)(ws + 3 * szQKV);
  float* comp = (float*)(ws + 4 * szQKV);                // 256*1024
  float* rw   = comp + (size_t)BB * NN * CC;             // 16384*16
  float* bm   = rw + (size_t)BB * TT * EE;               // 256
  float* rsn  = bm + BB * NN;                            // 32

  colnorm_kernel<<<1, 32, 0, stream>>>(gsim, fsim, rsn);
  compress_kernel<<<dim3(CC / 64, (BB * NN) / 64), 256, 0, stream>>>(hs, wc, bc, comp);
  gate_kernel<<<BB * NN, 256, 0, stream>>>(comp, gsim, rsn, ggat, thr, nullptr, bm, nullptr, 0);
  gate_kernel<<<BB * TT, 256, 0, stream>>>(hs, fsim, rsn + 16, fgat, thr, bm, nullptr, rw, 1);
  proj_kernel<<<dim3(EE, (BB * TT) / 64), 256, 0, stream>>>(hs, qp, pos, qb, 1);
  proj_kernel<<<dim3(EE, (BB * TT) / 64), 256, 0, stream>>>(hs, kp, pos, kb, 1);
  proj_kernel<<<dim3(EE, (BB * TT) / 64), 256, 0, stream>>>(hs, vp, pos, vb, 0);
  attn_kernel<<<BB * EE * NN, 256, 0, stream>>>(qb, kb, vb, rw, cw);
  out_kernel<<<dim3(CC / 64, (BB * TT) / 64), 256, 0, stream>>>(cw, op, (float*)d_out);
}
